// _Head_48979807043860
// MI455X (gfx1250) — matrix operand order repaired and checked by reading, not compiled, not run on hardware
//
#include <hip/hip_runtime.h>
#include <hip/hip_bf16.h>

// ---------------------------------------------------------------------------
// Fused causal single-head attention for MI455X (gfx1250, wave32, WMMA).
//   B=8, T=2048, C=1024, HS=128. ~22 GFLOP -> compute-bound: all matmuls on
//   v_wmma_f32_16x16x32_bf16 (f32 accumulate).
// Pipeline:
//   0) prep: x -> bf16 once; W -> bf16 transposed [HS,C] (contiguous B loads)
//   1) qkv: 8-wave blocks; each Wt K-chunk (8KB) staged into LDS once and
//      shared by 8 waves (8x less L2 traffic: 786MB -> 98MB); 8 WMMAs share
//      one A operand per chunk. V stored transposed [HS,T] for attention.
//   2) attn: flash-attention, one wave per 16-query tile, 32-key blocks.
//      All WMMA operands load as 2x b128 per lane. Online softmax via
//      shfl_xor row butterflies; P transposed C->A layout through LDS;
//      global_prefetch_b8 for the next key block.
// ---------------------------------------------------------------------------

typedef __bf16 bf16_t;
typedef __attribute__((ext_vector_type(16))) __bf16 v16bf;
typedef __attribute__((ext_vector_type(8)))  __bf16 bf16x8;
typedef __attribute__((ext_vector_type(8)))  float  v8f;

#define BATCH 8
#define TSEQ  2048
#define CDIM  1024
#define HS    128
#define PSTRIDE 40   // LDS row stride (bf16): 80B rows, 16B aligned, no 64-bank conflicts
#define WSTRIDE 40

#define CAT16(lo, hi) __builtin_shufflevector(lo, hi, 0,1,2,3,4,5,6,7,8,9,10,11,12,13,14,15)

// ------------------------------ prep kernels -------------------------------
__global__ __launch_bounds__(256)
void prep_x_kernel(const float* __restrict__ x, bf16_t* __restrict__ xb) {
    const size_t i = ((size_t)blockIdx.x * 256 + threadIdx.x) * 8;
    const float4 f0 = *(const float4*)(x + i);
    const float4 f1 = *(const float4*)(x + i + 4);
    bf16x8 o;
    o[0] = (bf16_t)f0.x; o[1] = (bf16_t)f0.y; o[2] = (bf16_t)f0.z; o[3] = (bf16_t)f0.w;
    o[4] = (bf16_t)f1.x; o[5] = (bf16_t)f1.y; o[6] = (bf16_t)f1.z; o[7] = (bf16_t)f1.w;
    *(bf16x8*)(xb + i) = o;
}

// W [C,HS] f32 -> Wt [HS,C] bf16 (x3 matrices via blockIdx.y)
__global__ __launch_bounds__(256)
void prep_w_kernel(const float* __restrict__ wq, const float* __restrict__ wk,
                   const float* __restrict__ wv, bf16_t* __restrict__ wt) {
    const int z = blockIdx.y;
    const float* w = (z == 0) ? wq : (z == 1) ? wk : wv;
    bf16_t* dst = wt + (size_t)z * CDIM * HS;
    const int idx = blockIdx.x * 256 + threadIdx.x;   // over C*HS = 131072
    const int row = idx / HS, col = idx % HS;
    dst[(size_t)col * CDIM + row] = (bf16_t)w[idx];
}

// -------------------------------- QKV projection ---------------------------
// grid: (B*T/128, 3), block 256 (8 waves). Wave w owns 16-row x-tile; the
// 8KB Wt chunk per K-step is staged to LDS once and shared by all 8 waves.
__global__ __launch_bounds__(256)
void qkv_proj_kernel(const bf16_t* __restrict__ xb,
                     const bf16_t* __restrict__ wt,
                     bf16_t* __restrict__ q,
                     bf16_t* __restrict__ k,
                     bf16_t* __restrict__ vt)
{
    __shared__ bf16_t w_lds[HS * WSTRIDE];   // 128 rows x 40 bf16 = 10 KB

    const int tid  = threadIdx.x;
    const int wave = tid >> 5;
    const int lane = tid & 31;
    const int h    = lane >> 4;
    const int l16  = lane & 15;
    const int rowBase = (blockIdx.x * 8 + wave) * 16;     // over B*T
    const int z       = blockIdx.y;
    const bf16_t* wtm = wt + (size_t)z * CDIM * HS;

    v8f acc[8];
#pragma unroll
    for (int t = 0; t < 8; ++t) acc[t] = (v8f){};

    const bf16_t* xrow = xb + (size_t)(rowBase + l16) * CDIM;
    for (int c = 0; c < CDIM; c += 32) {
        // ---- cooperative stage: Wt[0:128, c:c+32] -> LDS (512 x 16B) ----
#pragma unroll
        for (int j = 0; j < 2; ++j) {
            const int ch  = tid + j * 256;       // 0..511
            const int row = ch >> 2, q4 = ch & 3;
            const bf16x8 d = *(const bf16x8*)(wtm + (size_t)row * CDIM + c + q4 * 8);
            *(bf16x8*)(&w_lds[row * WSTRIDE + q4 * 8]) = d;
        }
        __syncthreads();

        // A (16x32): per-lane K = h*16+0..15 contiguous -> two b128 loads
        const bf16x8 alo = *(const bf16x8*)(xrow + c + h * 16);
        const bf16x8 ahi = *(const bf16x8*)(xrow + c + h * 16 + 8);
        const v16bf a = CAT16(alo, ahi);
#pragma unroll
        for (int nt = 0; nt < 8; ++nt) {
            // B (32x16): lane = col n; K = h*16+0..15 contiguous in LDS row n
            const bf16_t* wrow = &w_lds[(nt * 16 + l16) * WSTRIDE + h * 16];
            const v16bf bm = CAT16(*(const bf16x8*)wrow, *(const bf16x8*)(wrow + 8));
            acc[nt] = __builtin_amdgcn_wmma_f32_16x16x32_bf16(
                          false, a, false, bm, (short)0, acc[nt], false, false);
        }
        __syncthreads();
    }

    if (z < 2) {  // Q / K row-major [B*T, HS]
        bf16_t* dst = (z == 0) ? q : k;
#pragma unroll
        for (int nt = 0; nt < 8; ++nt)
#pragma unroll
            for (int r = 0; r < 8; ++r)
                dst[(size_t)(rowBase + r + 8 * h) * HS + nt * 16 + l16] =
                    (bf16_t)acc[nt][r];
    } else {      // V transposed per batch: vt[b][col][t]
        const int b    = rowBase / TSEQ;
        const int trow = rowBase % TSEQ;
        bf16_t* dst = vt + (size_t)b * HS * TSEQ;
#pragma unroll
        for (int nt = 0; nt < 8; ++nt)
#pragma unroll
            for (int r = 0; r < 8; ++r)
                dst[(size_t)(nt * 16 + l16) * TSEQ + trow + r + 8 * h] =
                    (bf16_t)acc[nt][r];
    }
}

// -------------------------------- Flash attention --------------------------
__global__ __launch_bounds__(32)
void attn_kernel(const bf16_t* __restrict__ qg,
                 const bf16_t* __restrict__ kg,
                 const bf16_t* __restrict__ vtg,
                 float* __restrict__ out)
{
    __shared__ bf16_t p_lds[16 * PSTRIDE];

    const int lane  = threadIdx.x & 31;
    const int h     = lane >> 4;
    const int l16   = lane & 15;
    const int b     = blockIdx.x / (TSEQ / 16);
    const int qbase = (blockIdx.x % (TSEQ / 16)) * 16;

    const bf16_t* qp  = qg  + (size_t)b * TSEQ * HS;
    const bf16_t* kp  = kg  + (size_t)b * TSEQ * HS;
    const bf16_t* vtp = vtg + (size_t)b * HS * TSEQ;

    // Q tile (16x128): four A operands, each two b128 loads per lane.
    v16bf aq[4];
    {
        const bf16_t* qrow = qp + (size_t)(qbase + l16) * HS;
#pragma unroll
        for (int c = 0; c < 4; ++c) {
            const bf16x8 lo = *(const bf16x8*)(qrow + c * 32 + h * 16);
            const bf16x8 hi = *(const bf16x8*)(qrow + c * 32 + h * 16 + 8);
            aq[c] = CAT16(lo, hi);
        }
    }

    v8f o[8];
#pragma unroll
    for (int t = 0; t < 8; ++t) o[t] = (v8f){};
    float mrow[8], lrow[8];
#pragma unroll
    for (int r = 0; r < 8; ++r) { mrow[r] = -__builtin_inff(); lrow[r] = 0.0f; }

    const float scale = 0.03125f;             // 1024^-0.5, exact
    const int nkb = (qbase + 15) / 32 + 1;    // causal: key0 <= qbase always

    for (int kb = 0; kb < nkb; ++kb) {
        const int key0 = kb * 32;

        // Prefetch next key block (global_prefetch_b8): 32 K rows + 128 Vt rows
        if (kb + 1 < nkb) {
            const int nx = key0 + 32;
            __builtin_prefetch(kp + (size_t)(nx + lane) * HS, 0, 3);
#pragma unroll
            for (int r = 0; r < 4; ++r)
                __builtin_prefetch(vtp + (size_t)(lane * 4 + r) * TSEQ + nx, 0, 3);
        }

        // ---- S = Q K^T (two 16x16 tiles covering 32 keys) ----
        v8f s[2];
#pragma unroll
        for (int st = 0; st < 2; ++st) {
            v8f accs = {};
            const bf16_t* krow = kp + (size_t)(key0 + st * 16 + l16) * HS;
#pragma unroll
            for (int c = 0; c < 4; ++c) {
                const bf16x8 lo = *(const bf16x8*)(krow + c * 32 + h * 16);
                const bf16x8 hi = *(const bf16x8*)(krow + c * 32 + h * 16 + 8);
                const v16bf bm = CAT16(lo, hi);
                accs = __builtin_amdgcn_wmma_f32_16x16x32_bf16(
                           false, aq[c], false, bm, (short)0, accs, false, false);
            }
            s[st] = accs;
        }

        // ---- scale + causal mask (C layout: row = r + 8h, col = l16) ----
#pragma unroll
        for (int st = 0; st < 2; ++st)
#pragma unroll
            for (int r = 0; r < 8; ++r) {
                const int qrow = qbase + r + 8 * h;
                const int key  = key0 + st * 16 + l16;
                const float vsc = s[st][r] * scale;
                s[st][r] = (key <= qrow) ? vsc : -__builtin_inff();
            }

        // ---- online softmax: row reductions across each 16-lane half ----
        float alpha[8];
#pragma unroll
        for (int r = 0; r < 8; ++r) {
            float mx = fmaxf(s[0][r], s[1][r]);
#pragma unroll
            for (int off = 8; off >= 1; off >>= 1)
                mx = fmaxf(mx, __shfl_xor(mx, off, 32));
            const float mnew = fmaxf(mrow[r], mx);
            alpha[r] = __expf(mrow[r] - mnew);   // exp(-inf)=0 on first block
            mrow[r]  = mnew;
            const float p0 = __expf(s[0][r] - mnew);
            const float p1 = __expf(s[1][r] - mnew);
            s[0][r] = p0; s[1][r] = p1;
            float rs = p0 + p1;
#pragma unroll
            for (int off = 8; off >= 1; off >>= 1)
                rs += __shfl_xor(rs, off, 32);
            lrow[r] = lrow[r] * alpha[r] + rs;
        }

        // ---- P (C layout) -> LDS -> reload in A layout (16x32 bf16) ----
#pragma unroll
        for (int st = 0; st < 2; ++st)
#pragma unroll
            for (int r = 0; r < 8; ++r)
                p_lds[(r + 8 * h) * PSTRIDE + st * 16 + l16] = (bf16_t)s[st][r];
        __syncthreads();
        v16bf pA;
        {
            const bf16_t* pr = &p_lds[l16 * PSTRIDE];
            const bf16x8 lo = *(const bf16x8*)(pr + h * 16);       // ds b128
            const bf16x8 hi = *(const bf16x8*)(pr + h * 16 + 8);   // ds b128
            pA = CAT16(lo, hi);
        }
        __syncthreads();

        // ---- rescale O, then O += P * V across 8 HS tiles ----
#pragma unroll
        for (int t = 0; t < 8; ++t)
#pragma unroll
            for (int r = 0; r < 8; ++r) o[t][r] *= alpha[r];

#pragma unroll
        for (int t = 0; t < 8; ++t) {
            // B = V chunk: element(kk,n) = Vt[t*16+n, key0+kk] -> contiguous
            const bf16_t* vrow = vtp + (size_t)(t * 16 + l16) * TSEQ + key0;
            const bf16x8 lo = *(const bf16x8*)(vrow + h * 16);
            const bf16x8 hi = *(const bf16x8*)(vrow + h * 16 + 8);
            const v16bf bv = CAT16(lo, hi);
            o[t] = __builtin_amdgcn_wmma_f32_16x16x32_bf16(
                       false, pA, false, bv, (short)0, o[t], false, false);
        }
    }

    // ---- epilogue: divide by softmax denominator, store f32 ----
#pragma unroll
    for (int r = 0; r < 8; ++r) lrow[r] = 1.0f / lrow[r];
    float* orow = out + ((size_t)b * TSEQ + qbase) * HS;
#pragma unroll
    for (int t = 0; t < 8; ++t)
#pragma unroll
        for (int r = 0; r < 8; ++r)
            orow[(size_t)(r + 8 * h) * HS + t * 16 + l16] = o[t][r] * lrow[r];
}

// -------------------------------- launcher ---------------------------------
extern "C" void kernel_launch(void* const* d_in, const int* in_sizes, int n_in,
                              void* d_out, int out_size, void* d_ws, size_t ws_size,
                              hipStream_t stream) {
    const float* x  = (const float*)d_in[0];
    const float* Wq = (const float*)d_in[1];
    const float* Wk = (const float*)d_in[2];
    const float* Wv = (const float*)d_in[3];
    float* out = (float*)d_out;

    const size_t NX = (size_t)BATCH * TSEQ * CDIM;  // 16,777,216
    const size_t NQ = (size_t)BATCH * TSEQ * HS;    //  2,097,152
    const size_t NW = (size_t)CDIM * HS;            //    131,072

    bf16_t* xb = (bf16_t*)d_ws;        // ws layout (~45 MB total):
    bf16_t* qb = xb + NX;              //   xb | q | k | vt | wt(x3)
    bf16_t* kb = qb + NQ;
    bf16_t* vt = kb + NQ;
    bf16_t* wt = vt + NQ;

    prep_x_kernel<<<dim3(NX / (256 * 8)), 256, 0, stream>>>(x, xb);
    prep_w_kernel<<<dim3(NW / 256, 3), 256, 0, stream>>>(Wq, Wk, Wv, wt);

    dim3 g1((BATCH * TSEQ) / (16 * 8), 3);
    qkv_proj_kernel<<<g1, 256, 0, stream>>>(xb, wt, qb, kb, vt);

    dim3 g2(BATCH * (TSEQ / 16));
    attn_kernel<<<g2, 32, 0, stream>>>(qb, kb, vt, out);
}
